// Inception_87729001988954
// MI455X (gfx1250) — matrix operand order repaired and checked by reading, not compiled, not run on hardware
//
#include <hip/hip_runtime.h>
#include <hip/hip_bf16.h>
#include <stdint.h>

// ---------------- problem constants ----------------
#define BATCH 4096
#define FEAT  10
#define TLEN  480
#define NWIN  24
#define WLEN  20
#define HID   64
#define NPAIR 45
#define DIN   210
#define DPAD  224            // 210 padded to 7*32 for f16 WMMA K-chunks
#define G3    192            // 3*HID
#define NT    12             // 192/16 N-tiles

typedef __attribute__((ext_vector_type(16))) _Float16 v16h;
typedef __attribute__((ext_vector_type(8)))  _Float16 v8h;
typedef __attribute__((ext_vector_type(8)))  float    v8f;

union AFrag { v16h v; v8h h[2]; };

__device__ __forceinline__ v8f wmma_f16(v16h a, v16h b, v8f c) {
    // D = A(16x32 f16) x B(32x16 f16) + C(16x16 f32)
    return __builtin_amdgcn_wmma_f32_16x16x32_f16(false, a, false, b, (short)0, c, false, false);
}
__device__ __forceinline__ float sigmoidf_(float x) { return 1.0f / (1.0f + __expf(-x)); }

// ---------------- workspace layout (bytes) ----------------
static constexpr size_t OFF_SRAW  = 0;                                        // f32 (b,w,224) -- aliased by gi
static constexpr size_t OFF_X     = (size_t)BATCH * NWIN * DPAD * 4;          // f16 (t*B+b, 224)
static constexpr size_t OFF_WIH0P = OFF_X + (size_t)BATCH * NWIN * DPAD * 2;  // f16 192x224
static constexpr size_t OFF_WHH0  = OFF_WIH0P + (size_t)G3 * DPAD * 2;        // f16 192x64
static constexpr size_t OFF_WIH1  = OFF_WHH0 + (size_t)G3 * HID * 2;
static constexpr size_t OFF_WHH1  = OFF_WIH1 + (size_t)G3 * HID * 2;
static constexpr size_t OFF_BIASA = OFF_WHH1 + (size_t)G3 * HID * 2;          // f32 192
static constexpr size_t OFF_BN0   = OFF_BIASA + 192 * 4;                      // f32 64
static constexpr size_t OFF_BRZ1  = OFF_BN0 + 64 * 4;                         // f32 128
static constexpr size_t OFF_BIN1  = OFF_BRZ1 + 128 * 4;                      // f32 64
static constexpr size_t OFF_BHN1  = OFF_BIN1 + 64 * 4;                       // f32 64
static constexpr size_t OFF_PART  = OFF_BHN1 + 64 * 4;                       // f32 4096x28
static constexpr size_t OFF_GSUM  = OFF_PART + (size_t)BATCH * 28 * 4;       // f32 28
static constexpr size_t OFF_SCALE = OFF_GSUM + 32 * 4;                       // f32 14
static constexpr size_t OFF_SHIFT = OFF_SCALE + 16 * 4;                      // f32 14

// =====================================================================
// Phase 1: per-batch window statistics + per-block BN partial sums
// =====================================================================
__global__ __launch_bounds__(256) void stats_kernel(
    const float* __restrict__ data, const int* __restrict__ comb_a,
    const int* __restrict__ comb_b, float* __restrict__ s_raw,
    float* __restrict__ partials)
{
    __shared__ float sx[FEAT * TLEN];     // 4800 f32
    __shared__ float smean[FEAT * NWIN];  // 240
    __shared__ float sstd[FEAT * NWIN];
    __shared__ float red[8];

    const int b = blockIdx.x, tid = threadIdx.x;
    for (int i = tid; i < FEAT * TLEN; i += 256) sx[i] = data[(size_t)b * FEAT * TLEN + i];
    __syncthreads();

    float gs[14], gq[14];
#pragma unroll
    for (int g = 0; g < 14; g++) { gs[g] = 0.f; gq[g] = 0.f; }

    if (tid < FEAT * NWIN) {
        const int f = tid / NWIN, w = tid % NWIN;
        const float* xb = &sx[f * TLEN + w * WLEN];
        float v[WLEN];
#pragma unroll
        for (int i = 0; i < WLEN; i++) v[i] = xb[i];

        float sum = 0.f, mn = v[0], wsum = 0.f;
#pragma unroll
        for (int i = 0; i < WLEN; i++) {
            sum += v[i]; mn = fminf(mn, v[i]); wsum += v[i] * (float)(i + 1);
        }
        wsum *= (1.0f / 210.0f);
        const float mean = sum * (1.0f / (float)WLEN);
        float m2 = 0.f, m3 = 0.f, m4 = 0.f;
#pragma unroll
        for (int i = 0; i < WLEN; i++) {
            float d = v[i] - mean, d2 = d * d;
            m2 += d2; m3 += d2 * d; m4 += d2 * d2;
        }
        m2 *= (1.0f / WLEN); m3 *= (1.0f / WLEN); m4 *= (1.0f / WLEN);
        const float sd = sqrtf(m2);
        const float ratio = v[WLEN - 1] / (v[0] + 0.01f) - 1.0f;
        // median via stable ranks (exact multiset order; fully unrolled -> registers)
        float mlo = 0.f, mhi = 0.f;
#pragma unroll
        for (int i = 0; i < WLEN; i++) {
            int c = 0;
#pragma unroll
            for (int j = 0; j < WLEN; j++)
                c += ((v[j] < v[i]) || (v[j] == v[i] && j < i)) ? 1 : 0;
            if (c == 9)  mlo = v[i];
            if (c == 10) mhi = v[i];
        }
        const float med = 0.5f * (mlo + mhi);

        float* sr = s_raw + ((size_t)b * NWIN + w) * DPAD;
#define EMIT(g, val) { float _v = (val); sr[(g)*10 + 70 + f] = _v; gs[g] += _v; gq[g] += _v * _v; }
        EMIT(2, sd)
        EMIT(3, mean / (sd + 0.01f))
        EMIT(4, ratio)
        EMIT(5, wsum)
        EMIT(6, mean)
        EMIT(7, mn)
        EMIT(8, m2)
        EMIT(9, sum)
        EMIT(10, med)
        EMIT(11, m3 / (sd * sd * sd + 0.01f))
        EMIT(12, m4 / (m2 * m2 + 0.01f) - 3.0f)
        EMIT(13, sum)
#undef EMIT
        smean[tid] = mean; sstd[tid] = sd;
    }
    __syncthreads();

    for (int task = tid; task < NPAIR * NWIN; task += 256) {
        const int p = task / NWIN, w = task % NWIN;
        const int fa = comb_a[p], fb = comb_b[p];
        const float ma = smean[fa * NWIN + w], mb = smean[fb * NWIN + w];
        const float sa = sstd[fa * NWIN + w], sb = sstd[fb * NWIN + w];
        const float* xa = &sx[fa * TLEN + w * WLEN];
        const float* xc = &sx[fb * TLEN + w * WLEN];
        float dot = 0.f;
#pragma unroll
        for (int i = 0; i < WLEN; i++) dot += xa[i] * xc[i];
        const float cov  = (dot - (float)WLEN * ma * mb) * (1.0f / (WLEN - 1));
        const float corr = cov / (sa * sb + 0.01f);
        float* sr = s_raw + ((size_t)b * NWIN + w) * DPAD;
        sr[p] = corr; sr[NPAIR + p] = cov;
        gs[0] += corr; gq[0] += corr * corr;
        gs[1] += cov;  gq[1] += cov * cov;
    }

    // deterministic block reduction of 28 quantities (14 sums + 14 sumsq)
    const int lane = tid & 31, wid = tid >> 5;
#pragma unroll
    for (int q = 0; q < 28; q++) {
        float val = (q < 14) ? gs[q] : gq[q - 14];
#pragma unroll
        for (int off = 16; off > 0; off >>= 1) val += __shfl_down(val, off, 32);
        if (lane == 0) red[wid] = val;
        __syncthreads();
        if (tid == 0) {
            float s = 0.f;
#pragma unroll
            for (int wv = 0; wv < 8; wv++) s += red[wv];
            partials[(size_t)b * 28 + q] = s;
        }
        __syncthreads();
    }
}

// =====================================================================
// Phase 2: deterministic tree reduction of per-block partials
// =====================================================================
__global__ __launch_bounds__(256) void reduce_kernel(const float* __restrict__ partials,
                                                     float* __restrict__ gsums)
{
    __shared__ float red[256];
    const int q = blockIdx.x, tid = threadIdx.x;
    float s = 0.f;
    for (int i = tid; i < BATCH; i += 256) s += partials[(size_t)i * 28 + q];
    red[tid] = s;
    __syncthreads();
    for (int o = 128; o > 0; o >>= 1) {
        if (tid < o) red[tid] += red[tid + o];
        __syncthreads();
    }
    if (tid == 0) gsums[q] = red[0];
}

// Phase 2b: per-group batchnorm affine params (fold conv + BN into scale/shift)
__global__ void bnparam_kernel(const float* __restrict__ gsums,
                               const float* __restrict__ conv_w, const float* __restrict__ conv_b,
                               const float* __restrict__ gamma, const float* __restrict__ beta,
                               float* __restrict__ scale, float* __restrict__ shift)
{
    const int g = threadIdx.x;
    if (g < 14) {
        const float cw = conv_w[0], cb = conv_b[0];
        const float Ng = (float)((size_t)BATCH * NWIN * (g < 2 ? NPAIR : FEAT));
        const float m   = gsums[g] / Ng;
        const float var = gsums[14 + g] / Ng - m * m;
        const float my = cw * m + cb;
        const float vy = cw * cw * var;
        const float rs = rsqrtf(vy + 1e-5f);
        scale[g] = gamma[g] * cw * rs;
        shift[g] = gamma[g] * rs * (cb - my) + beta[g];
    }
}

// =====================================================================
// Phase 3a: weight/bias prep (f32 -> f16, K-pad, bias folding)
// =====================================================================
__global__ __launch_bounds__(256) void wprep_kernel(
    const float* __restrict__ Wih0, const float* __restrict__ Whh0,
    const float* __restrict__ bih0, const float* __restrict__ bhh0,
    const float* __restrict__ Wih1, const float* __restrict__ Whh1,
    const float* __restrict__ bih1, const float* __restrict__ bhh1,
    _Float16* __restrict__ Wih0p, _Float16* __restrict__ Whh0h,
    _Float16* __restrict__ Wih1h, _Float16* __restrict__ Whh1h,
    float* __restrict__ biasA, float* __restrict__ bN0, float* __restrict__ bRZ1,
    float* __restrict__ bIN1, float* __restrict__ bHN1)
{
    int idx = blockIdx.x * 256 + threadIdx.x;
    const int S0 = G3 * DPAD;        // 43008
    const int S1 = S0 + G3 * HID;    // +12288
    const int S2 = S1 + G3 * HID;
    const int S3 = S2 + G3 * HID;
    const int S4 = S3 + G3;          // biasA
    const int S5 = S4 + HID;         // bN0
    const int S6 = S5 + 2 * HID;     // bRZ1
    const int S7 = S6 + HID;         // bIN1
    const int S8 = S7 + HID;         // bHN1
    if (idx < S0) {
        int r = idx / DPAD, c = idx % DPAD;
        Wih0p[idx] = (c < DIN) ? (_Float16)Wih0[r * DIN + c] : (_Float16)0.0f;
    } else if (idx < S1) { int i = idx - S0; Whh0h[i] = (_Float16)Whh0[i]; }
    else if (idx < S2)   { int i = idx - S1; Wih1h[i] = (_Float16)Wih1[i]; }
    else if (idx < S3)   { int i = idx - S2; Whh1h[i] = (_Float16)Whh1[i]; }
    else if (idx < S4)   { int i = idx - S3; biasA[i] = bih0[i] + (i < 2 * HID ? bhh0[i] : 0.0f); }
    else if (idx < S5)   { int i = idx - S4; bN0[i]  = bhh0[2 * HID + i]; }
    else if (idx < S6)   { int i = idx - S5; bRZ1[i] = bih1[i] + bhh1[i]; }
    else if (idx < S7)   { int i = idx - S6; bIN1[i] = bih1[2 * HID + i]; }
    else if (idx < S8)   { int i = idx - S7; bHN1[i] = bhh1[2 * HID + i]; }
}

// =====================================================================
// Phase 3b: normalize stats -> f16 GRU input X, laid out (t*B + b, 224)
// =====================================================================
__global__ __launch_bounds__(256) void xbuild_kernel(
    const float* __restrict__ s_raw, const float* __restrict__ scale,
    const float* __restrict__ shift, _Float16* __restrict__ X)
{
    const size_t N = (size_t)BATCH * NWIN * DPAD;
    const size_t stride = (size_t)gridDim.x * blockDim.x;
    for (size_t n = (size_t)blockIdx.x * blockDim.x + threadIdx.x; n < N; n += stride) {
        const int ch = (int)(n % DPAD);
        const size_t row = n / DPAD;           // row = t*BATCH + b
        _Float16 val = (_Float16)0.0f;
        if (ch < DIN) {
            const int t = (int)(row / BATCH), bb = (int)(row % BATCH);
            const int g = (ch < 45) ? 0 : (ch < 90) ? 1 : (2 + (ch - 90) / 10);
            const float s = s_raw[((size_t)bb * NWIN + t) * DPAD + ch];
            val = (_Float16)(scale[g] * s + shift[g]);
        }
        X[n] = val;
    }
}

// =====================================================================
// Phase 4: Gi0 = X @ Wih0^T + folded biases, stored in WMMA C-layout.
// M = 98304 (t-major), K = 224, N = 192. 8 waves/block, 1 M-tile/wave.
// =====================================================================
__global__ __launch_bounds__(256) void gemm0_kernel(
    const _Float16* __restrict__ X, const _Float16* __restrict__ Wp,
    const float* __restrict__ biasA, float* __restrict__ gi)
{
    const int wave = threadIdx.x >> 5, lane = threadIdx.x & 31;
    const int half = lane >> 4, mr = lane & 15;
    const int mt = blockIdx.x * 8 + wave;

    const _Float16* xr = X + (size_t)(mt * 16 + mr) * DPAD;
    AFrag a[7];
#pragma unroll
    for (int kc = 0; kc < 7; kc++) {
        a[kc].h[0] = *(const v8h*)(xr + kc * 32 + half * 16);
        a[kc].h[1] = *(const v8h*)(xr + kc * 32 + half * 16 + 8);
    }
#pragma unroll
    for (int tn = 0; tn < NT; tn++) {
        const int col = tn * 16 + mr;
        const float bv = biasA[col];
        v8f acc;
#pragma unroll
        for (int v = 0; v < 8; v++) acc[v] = bv;
        const _Float16* wr = Wp + (size_t)col * DPAD;
#pragma unroll
        for (int kc = 0; kc < 7; kc++) {
            v16h bf = *(const v16h*)(wr + kc * 32 + half * 16);
            acc = wmma_f16(a[kc].v, bf, acc);
        }
        *(v8f*)(gi + ((size_t)(mt * NT + tn) * 32 + lane) * 8) = acc;
    }
}

// =====================================================================
// Phase 5: fused 2-layer GRU scan. 1 wave = 16 batch rows; h-state kept
// in C-layout registers + f16 LDS mirror for the WMMA A operand.
// =====================================================================
__global__ __launch_bounds__(128) void gru_kernel(
    const float* __restrict__ gi,
    const _Float16* __restrict__ Whh0, const _Float16* __restrict__ Wih1,
    const _Float16* __restrict__ Whh1,
    const float* __restrict__ bN0, const float* __restrict__ bRZ1,
    const float* __restrict__ bIN1, const float* __restrict__ bHN1,
    float* __restrict__ out)
{
    __shared__ __align__(32) _Float16 hbuf[4][2][16 * HID];   // 16 KB

    const int tid = threadIdx.x, wid = tid >> 5, lane = tid & 31;
    const int half = lane >> 4, mr = lane & 15;
    const int wt = blockIdx.x * 4 + wid;           // batch tile: rows wt*16 .. +15

    _Float16* hb0 = hbuf[wid][0];
    _Float16* hb1 = hbuf[wid][1];
    for (int i = lane; i < (16 * HID) / 2; i += 32) {
        ((uint32_t*)hb0)[i] = 0u;
        ((uint32_t*)hb1)[i] = 0u;
    }
    float h0r[4][8] = {}, h1r[4][8] = {};

    float bn0q[4], brz[8], binq[4], bhnq[4];
#pragma unroll
    for (int q = 0; q < 4; q++) {
        bn0q[q] = bN0[q * 16 + mr];
        binq[q] = bIN1[q * 16 + mr];
        bhnq[q] = bHN1[q * 16 + mr];
    }
#pragma unroll
    for (int tn = 0; tn < 8; tn++) brz[tn] = bRZ1[tn * 16 + mr];

    for (int t = 0; t < NWIN; t++) {
        const float* gbase = gi + ((size_t)(t * (BATCH / 16) + wt) * NT * 32 + lane) * 8;

        // ---- layer 0: gh0 = h0 @ Whh0^T ----
        AFrag a0[2];
#pragma unroll
        for (int kc = 0; kc < 2; kc++) {
            a0[kc].h[0] = *(const v8h*)(hb0 + mr * HID + kc * 32 + half * 16);
            a0[kc].h[1] = *(const v8h*)(hb0 + mr * HID + kc * 32 + half * 16 + 8);
        }
        v8f acc[12];
#pragma unroll
        for (int tn = 0; tn < 8; tn++)
            acc[tn] = *(const v8f*)(gbase + (size_t)tn * 256);   // gi has bih+bhh folded (r,z)
#pragma unroll
        for (int q = 0; q < 4; q++) {
#pragma unroll
            for (int v = 0; v < 8; v++) acc[8 + q][v] = bn0q[q]; // gh_n seeded with bhh_n
        }
#pragma unroll
        for (int tn = 0; tn < 12; tn++) {
            const _Float16* wr = Whh0 + (size_t)(tn * 16 + mr) * HID;
#pragma unroll
            for (int kc = 0; kc < 2; kc++) {
                v16h bf = *(const v16h*)(wr + kc * 32 + half * 16);
                acc[tn] = wmma_f16(a0[kc].v, bf, acc[tn]);
            }
        }
        // ---- layer-0 gates ----
#pragma unroll
        for (int q = 0; q < 4; q++) {
            v8f gn = *(const v8f*)(gbase + (size_t)(8 + q) * 256);  // gi_n (bih only)
#pragma unroll
            for (int v = 0; v < 8; v++) {
                float r  = sigmoidf_(acc[q][v]);
                float z  = sigmoidf_(acc[4 + q][v]);
                float nn = tanhf(gn[v] + r * acc[8 + q][v]);
                h0r[q][v] = (1.0f - z) * nn + z * h0r[q][v];
            }
        }
#pragma unroll
        for (int q = 0; q < 4; q++)
#pragma unroll
            for (int v = 0; v < 8; v++)
                hb0[(v + 8 * half) * HID + q * 16 + mr] = (_Float16)h0r[q][v];

        // ---- layer 1: r/z share one accumulator (bias = bih1+bhh1) ----
        AFrag ay[2], ah[2];
#pragma unroll
        for (int kc = 0; kc < 2; kc++) {
            ay[kc].h[0] = *(const v8h*)(hb0 + mr * HID + kc * 32 + half * 16);
            ay[kc].h[1] = *(const v8h*)(hb0 + mr * HID + kc * 32 + half * 16 + 8);
            ah[kc].h[0] = *(const v8h*)(hb1 + mr * HID + kc * 32 + half * 16);
            ah[kc].h[1] = *(const v8h*)(hb1 + mr * HID + kc * 32 + half * 16 + 8);
        }
        v8f arz[8];
#pragma unroll
        for (int tn = 0; tn < 8; tn++) {
#pragma unroll
            for (int v = 0; v < 8; v++) arz[tn][v] = brz[tn];
            const _Float16* wi = Wih1 + (size_t)(tn * 16 + mr) * HID;
            const _Float16* wh = Whh1 + (size_t)(tn * 16 + mr) * HID;
#pragma unroll
            for (int kc = 0; kc < 2; kc++) {
                v16h bi = *(const v16h*)(wi + kc * 32 + half * 16);
                arz[tn] = wmma_f16(ay[kc].v, bi, arz[tn]);
                v16h bh = *(const v16h*)(wh + kc * 32 + half * 16);
                arz[tn] = wmma_f16(ah[kc].v, bh, arz[tn]);
            }
        }
        v8f ain[4], ahn[4];
#pragma unroll
        for (int q = 0; q < 4; q++) {
#pragma unroll
            for (int v = 0; v < 8; v++) { ain[q][v] = binq[q]; ahn[q][v] = bhnq[q]; }
            const _Float16* wi = Wih1 + (size_t)(128 + q * 16 + mr) * HID;
            const _Float16* wh = Whh1 + (size_t)(128 + q * 16 + mr) * HID;
#pragma unroll
            for (int kc = 0; kc < 2; kc++) {
                v16h bi = *(const v16h*)(wi + kc * 32 + half * 16);
                ain[q] = wmma_f16(ay[kc].v, bi, ain[q]);
                v16h bh = *(const v16h*)(wh + kc * 32 + half * 16);
                ahn[q] = wmma_f16(ah[kc].v, bh, ahn[q]);
            }
        }
#pragma unroll
        for (int q = 0; q < 4; q++) {
#pragma unroll
            for (int v = 0; v < 8; v++) {
                float r  = sigmoidf_(arz[q][v]);
                float z  = sigmoidf_(arz[4 + q][v]);
                float nn = tanhf(ain[q][v] + r * ahn[q][v]);
                h1r[q][v] = (1.0f - z) * nn + z * h1r[q][v];
            }
        }
#pragma unroll
        for (int q = 0; q < 4; q++)
#pragma unroll
            for (int v = 0; v < 8; v++)
                hb1[(v + 8 * half) * HID + q * 16 + mr] = (_Float16)h1r[q][v];
    }

    // final hidden state of layer 1 -> output (B, 64) f32
#pragma unroll
    for (int q = 0; q < 4; q++)
#pragma unroll
        for (int v = 0; v < 8; v++)
            out[(size_t)(wt * 16 + v + 8 * half) * HID + q * 16 + mr] = h1r[q][v];
}

// =====================================================================
extern "C" void kernel_launch(void* const* d_in, const int* in_sizes, int n_in,
                              void* d_out, int out_size, void* d_ws, size_t ws_size,
                              hipStream_t stream)
{
    const float* data   = (const float*)d_in[0];
    const float* conv_w = (const float*)d_in[1];
    const float* conv_b = (const float*)d_in[2];
    const float* gamma  = (const float*)d_in[3];
    const float* beta   = (const float*)d_in[4];
    const float* Wih0   = (const float*)d_in[5];
    const float* Whh0   = (const float*)d_in[6];
    const float* bih0   = (const float*)d_in[7];
    const float* bhh0   = (const float*)d_in[8];
    const float* Wih1   = (const float*)d_in[9];
    const float* Whh1   = (const float*)d_in[10];
    const float* bih1   = (const float*)d_in[11];
    const float* bhh1   = (const float*)d_in[12];
    const int*   comb_a = (const int*)d_in[13];
    const int*   comb_b = (const int*)d_in[14];

    char* ws = (char*)d_ws;
    float*    s_raw  = (float*)(ws + OFF_SRAW);
    float*    gi     = (float*)(ws + OFF_SRAW);   // aliases s_raw (dead after xbuild)
    _Float16* X      = (_Float16*)(ws + OFF_X);
    _Float16* Wih0p  = (_Float16*)(ws + OFF_WIH0P);
    _Float16* Whh0h  = (_Float16*)(ws + OFF_WHH0);
    _Float16* Wih1h  = (_Float16*)(ws + OFF_WIH1);
    _Float16* Whh1h  = (_Float16*)(ws + OFF_WHH1);
    float*    biasA  = (float*)(ws + OFF_BIASA);
    float*    bN0    = (float*)(ws + OFF_BN0);
    float*    bRZ1   = (float*)(ws + OFF_BRZ1);
    float*    bIN1   = (float*)(ws + OFF_BIN1);
    float*    bHN1   = (float*)(ws + OFF_BHN1);
    float*    part   = (float*)(ws + OFF_PART);
    float*    gsums  = (float*)(ws + OFF_GSUM);
    float*    scl    = (float*)(ws + OFF_SCALE);
    float*    shf    = (float*)(ws + OFF_SHIFT);

    stats_kernel<<<BATCH, 256, 0, stream>>>(data, comb_a, comb_b, s_raw, part);
    reduce_kernel<<<28, 256, 0, stream>>>(part, gsums);
    bnparam_kernel<<<1, 32, 0, stream>>>(gsums, conv_w, conv_b, gamma, beta, scl, shf);
    wprep_kernel<<<(G3 * DPAD + 3 * G3 * HID + G3 + 5 * HID + 2 * HID + 255) / 256, 256, 0, stream>>>(
        Wih0, Whh0, bih0, bhh0, Wih1, Whh1, bih1, bhh1,
        Wih0p, Whh0h, Wih1h, Whh1h, biasA, bN0, bRZ1, bIN1, bHN1);
    xbuild_kernel<<<4096, 256, 0, stream>>>(s_raw, scl, shf, X);
    gemm0_kernel<<<(BATCH * NWIN / 16) / 8, 256, 0, stream>>>(X, Wih0p, biasA, gi);
    gru_kernel<<<BATCH / 64, 128, 0, stream>>>(gi, Whh0h, Wih1h, Whh1h,
                                               bN0, bRZ1, bIN1, bHN1, (float*)d_out);
}
